// AudioAttentionMapGenerator_46076409151553
// MI455X (gfx1250) — matrix operand order repaired and checked by reading, not compiled, not run on hardware
//
#include <hip/hip_runtime.h>
#include <hip/hip_bf16.h>

typedef __attribute__((ext_vector_type(16))) _Float16 v16h;
typedef __attribute__((ext_vector_type(8)))  _Float16 v8h;
typedef __attribute__((ext_vector_type(8)))  float    v8f;

#define BB   16
#define NPTS 128
#define HH   224
#define WW   224

// ---------------------------------------------------------------------------
// Kernel 1: build separable Gaussian factor matrices in f16.
//   Afac[b][h][n]  = mask*weight[b,n] * exp(-c*(h - v_n)^2)   (k-contiguous)
//   Btfac[b][w][n] =                    exp(-c*(w - u_n)^2)   (k-contiguous)
// Also zero-initializes the per-batch max slots (ws is poisoned by harness).
// ---------------------------------------------------------------------------
__global__ void gauss_factors_kernel(const float* __restrict__ coords,   // (B,N,2)
                                     const float* __restrict__ wts,     // (B,N)
                                     const unsigned char* __restrict__ mask, // (B,N) bool
                                     const float* __restrict__ log_sigma,
                                     _Float16* __restrict__ Afac,
                                     _Float16* __restrict__ Btfac,
                                     unsigned int* __restrict__ bmax) {
    int idx = blockIdx.x * blockDim.x + threadIdx.x;
    if (idx < BB) bmax[idx] = 0u;           // values are >= 0, so 0 is -inf here
    const int total = BB * (HH + WW) * NPTS;
    if (idx >= total) return;

    const int n   = idx & (NPTS - 1);
    const int pos = (idx >> 7) % (HH + WW);
    const int b   = idx / ((HH + WW) * NPTS);

    const float sigma = __expf(log_sigma[0]);
    const float inv_two_var = 0.5f / (sigma * sigma + 1e-6f);

    const bool isA = (pos < HH);
    const float p  = isA ? (float)pos : (float)(pos - HH);
    const float cc = coords[(((b << 7) + n) << 1) + (isA ? 1 : 0)]; // v for A, u for B
    const float d  = p - cc;
    const float g  = __expf(-inv_two_var * d * d);

    if (isA) {
        const float wv = wts[(b << 7) + n] * (mask[(b << 7) + n] ? 1.0f : 0.0f);
        Afac[((size_t)b * HH + pos) * NPTS + n] = (_Float16)(g * wv);
    } else {
        Btfac[((size_t)b * WW + (pos - HH)) * NPTS + n] = (_Float16)g;
    }
}

// ---------------------------------------------------------------------------
// Kernel 2: batched GEMM  map[b] = Afac[b] (224x128) x Bfac[b] (128x224)
// via v_wmma_f32_16x16x32_f16. One wave32 per 16x16 output tile, 8 waves
// per block, K loop = 4 WMMA steps. Writes raw map to d_out and reduces the
// per-batch max into bmax (uint bit-pattern atomicMax; all values >= 0).
// ---------------------------------------------------------------------------
__global__ void wmma_map_kernel(const _Float16* __restrict__ Afac,
                                const _Float16* __restrict__ Btfac,
                                float* __restrict__ out,
                                unsigned int* __restrict__ bmax) {
    const int lane = threadIdx.x & 31;
    const int wave = threadIdx.x >> 5;
    const int tile = blockIdx.x * 8 + wave;          // 3136 tiles total, exact fit

    const int tilesPerBatch = (HH / 16) * (WW / 16); // 196
    const int b  = tile / tilesPerBatch;
    const int t  = tile % tilesPerBatch;
    const int tm = t / (WW / 16);
    const int tn = t % (WW / 16);

    const int half = lane >> 4;   // 0: lanes 0-15, 1: lanes 16-31
    const int r    = lane & 15;

    const _Float16* Ab = Afac  + ((size_t)b * HH + tm * 16) * NPTS;
    const _Float16* Bb = Btfac + ((size_t)b * WW + tn * 16) * NPTS;

    v8f c = {};
#pragma unroll
    for (int k0 = 0; k0 < NPTS; k0 += 32) {
        // A fragment (16x32, row M = r): lane half selects K+16 block;
        // elements 0..15 -> K = k0+half*16+0..15.
        const _Float16* aptr = Ab + r * NPTS + k0 + half * 16;
        v8h alo = *(const v8h*)(aptr);
        v8h ahi = *(const v8h*)(aptr + 8);
        v16h a;
#pragma unroll
        for (int i = 0; i < 8; ++i) { a[i] = alo[i]; a[i + 8] = ahi[i]; }

        // B fragment (32x16, col N = r): lane half selects K+16 block;
        // elements 0..15 -> K = k0+half*16+0..15 (contiguous in Bt layout).
        const _Float16* bptr = Bb + r * NPTS + k0 + half * 16;
        v16h bfrag = *(const v16h*)(bptr);

        c = __builtin_amdgcn_wmma_f32_16x16x32_f16(
                /*neg_a=*/false, a, /*neg_b=*/false, bfrag,
                /*c_mod=*/(short)0, c, /*reuse_a=*/false, /*reuse_b=*/false);
    }

    // C/D layout: VGPR rr -> row M = tm*16 + rr + half*8, col N = tn*16 + r.
    float m = 0.0f;
    const int ncol = tn * 16 + r;
#pragma unroll
    for (int rr = 0; rr < 8; ++rr) {
        const int mrow = tm * 16 + rr + half * 8;
        out[((size_t)b * HH + mrow) * WW + ncol] = c[rr];
        m = fmaxf(m, c[rr]);
    }
    // wave32 max reduction, then one atomic per wave
#pragma unroll
    for (int off = 16; off > 0; off >>= 1)
        m = fmaxf(m, __shfl_down(m, off, 32));
    if (lane == 0)
        atomicMax(bmax + b, __float_as_uint(m));
}

// ---------------------------------------------------------------------------
// Kernel 3: in-place normalization by (per-batch max + 1e-6).
// ---------------------------------------------------------------------------
__global__ void normalize_kernel(float* __restrict__ out,
                                 const unsigned int* __restrict__ bmax) {
    int idx = blockIdx.x * blockDim.x + threadIdx.x;
    const int total = BB * HH * WW;
    if (idx >= total) return;
    const int b = idx / (HH * WW);
    const float mx = __uint_as_float(bmax[b]);
    out[idx] = out[idx] / (mx + 1e-6f);
}

extern "C" void kernel_launch(void* const* d_in, const int* in_sizes, int n_in,
                              void* d_out, int out_size, void* d_ws, size_t ws_size,
                              hipStream_t stream) {
    const float*         coords    = (const float*)d_in[0];         // (B,N,2)
    const float*         wts       = (const float*)d_in[1];         // (B,N)
    const unsigned char* mask      = (const unsigned char*)d_in[2]; // (B,N) bool
    const float*         log_sigma = (const float*)d_in[3];         // scalar
    float* out = (float*)d_out;

    // Workspace layout (hipMalloc base is 256B aligned):
    //   [0, 64)                      : 16 x uint per-batch max
    //   [256, 256+917504)            : Afac  f16 (B*H*128)
    //   [917760, 917760+917504)      : Btfac f16 (B*W*128)  (32B aligned)
    unsigned int* bmax  = (unsigned int*)d_ws;
    _Float16*     Afac  = (_Float16*)((char*)d_ws + 256);
    _Float16*     Btfac = (_Float16*)((char*)d_ws + 256 + (size_t)BB * HH * NPTS * 2);

    // 1) Gaussian factor matrices (+ bmax init)
    {
        const int total = BB * (HH + WW) * NPTS;   // 917,504
        const int block = 256;
        gauss_factors_kernel<<<(total + block - 1) / block, block, 0, stream>>>(
            coords, wts, mask, log_sigma, Afac, Btfac, bmax);
    }
    // 2) Batched WMMA GEMM -> raw map + per-batch max
    {
        const int tiles = BB * (HH / 16) * (WW / 16);  // 3136, divisible by 8
        wmma_map_kernel<<<tiles / 8, 256, 0, stream>>>(Afac, Btfac, out, bmax);
    }
    // 3) Normalize in place
    {
        const int total = BB * HH * WW;
        const int block = 256;
        normalize_kernel<<<(total + block - 1) / block, block, 0, stream>>>(out, bmax);
    }
}
